// LongTermEncoder_85487029059678
// MI455X (gfx1250) — matrix operand order repaired and checked by reading, not compiled, not run on hardware
//
#include <hip/hip_runtime.h>
#include <hip/hip_bf16.h>

typedef _Float16 f16;
typedef __attribute__((ext_vector_type(16))) _Float16 v16h;
typedef __attribute__((ext_vector_type(8)))  _Float16 v8h;
typedef __attribute__((ext_vector_type(8)))  float    v8f;

#define NN    1000
#define NPAD  1024
#define NDIM  40
#define BB    8
#define RCH   8
#define CCH   32
#define TMAXX 161
#define GPA   0.05f
#define TKTOP 20

// ---------------- v1/v2 = tanh(3*(emb @ W^T + b)) ----------------
__global__ void k_v12(const float* __restrict__ emb1, const float* __restrict__ emb2,
                      const float* __restrict__ l1w, const float* __restrict__ l1b,
                      const float* __restrict__ l2w, const float* __restrict__ l2b,
                      float* __restrict__ v1, float* __restrict__ v2) {
  int gid = blockIdx.x * 256 + threadIdx.x;
  if (gid >= NN * NDIM) return;
  int i = gid / NDIM, j = gid % NDIM;
  float s1 = l1b[j], s2 = l2b[j];
  for (int k = 0; k < NDIM; ++k) {
    s1 += emb1[i * NDIM + k] * l1w[j * NDIM + k];
    s2 += emb2[i * NDIM + k] * l2w[j * NDIM + k];
  }
  v1[gid] = tanhf(3.0f * s1);
  v2[gid] = tanhf(3.0f * s2);
}

// ---------------- per-row adjacency: relu(tanh(3a)), top-20, softmax -> P ----------------
__global__ __launch_bounds__(256) void k_adj(const float* __restrict__ v1, const float* __restrict__ v2,
                                             const float* __restrict__ noise, const float* __restrict__ cool,
                                             float* __restrict__ P) {
  __shared__ float v1i[NDIM], v2i[NDIM];
  __shared__ float adjrow[NN];
  __shared__ float score[NN];
  __shared__ unsigned char keep[NN];
  __shared__ float sv[256];
  __shared__ int   si[256];
  __shared__ float smax, ssum;
  int i = blockIdx.x, tid = threadIdx.x;
  if (i >= NN) {                                  // padding rows of P are zero
    for (int j = tid; j < NPAD; j += 256) P[(size_t)i * NPAD + j] = 0.f;
    return;
  }
  for (int k = tid; k < NDIM; k += 256) { v1i[k] = v1[i * NDIM + k]; v2i[k] = v2[i * NDIM + k]; }
  __syncthreads();
  for (int j = tid; j < NN; j += 256) {
    float d1 = 0.f, d2 = 0.f;
    for (int k = 0; k < NDIM; ++k) {
      d1 += v1i[k] * v2[j * NDIM + k];
      d2 += v2i[k] * v1[j * NDIM + k];
    }
    float a = d1 - d2;
    float ad = tanhf(3.0f * a); ad = ad > 0.f ? ad : 0.f;
    adjrow[j] = ad;
    score[j] = ad + 0.01f * noise[(size_t)i * NN + j];
    keep[j] = 0;
  }
  __syncthreads();
  // iterative top-20 argmax, tie -> smaller index (matches lax.top_k set)
  for (int it = 0; it < TKTOP; ++it) {
    float bv = -1e30f; int bi = 0x7fffffff;
    for (int j = tid; j < NN; j += 256) {
      float s = score[j];
      if (s > bv || (s == bv && j < bi)) { bv = s; bi = j; }
    }
    sv[tid] = bv; si[tid] = bi; __syncthreads();
    for (int off = 128; off > 0; off >>= 1) {
      if (tid < off) {
        if (sv[tid + off] > sv[tid] || (sv[tid + off] == sv[tid] && si[tid + off] < si[tid])) {
          sv[tid] = sv[tid + off]; si[tid] = si[tid + off];
        }
      }
      __syncthreads();
    }
    if (tid == 0) { keep[si[0]] = 1; score[si[0]] = -1e30f; }
    __syncthreads();
  }
  // z = adj*mask*(mv_i*mv_j); softmax over full row (zeros included)
  float mvi = 1.f - cool[i];
  for (int j = tid; j < NN; j += 256)
    score[j] = keep[j] ? adjrow[j] * mvi * (1.f - cool[j]) : 0.f;
  __syncthreads();
  float lm = -1e30f;
  for (int j = tid; j < NN; j += 256) lm = fmaxf(lm, score[j]);
  sv[tid] = lm; __syncthreads();
  for (int off = 128; off > 0; off >>= 1) { if (tid < off) sv[tid] = fmaxf(sv[tid], sv[tid + off]); __syncthreads(); }
  if (tid == 0) smax = sv[0];
  __syncthreads();
  float ls = 0.f;
  for (int j = tid; j < NN; j += 256) ls += expf(score[j] - smax);
  sv[tid] = ls; __syncthreads();
  for (int off = 128; off > 0; off >>= 1) { if (tid < off) sv[tid] += sv[tid + off]; __syncthreads(); }
  if (tid == 0) ssum = sv[0];
  __syncthreads();
  float inv = 1.f / ssum;
  for (int j = tid; j < NPAD; j += 256) {
    float p = (j < NN) ? expf(score[j] - smax) * inv : 0.f;
    P[(size_t)i * NPAD + j] = p;
  }
}

// ---------------- mixprop normalization: A16=(P+I)/rowsum, At16=(P^T+I)/colsum (f16, padded) ----------------
__global__ __launch_bounds__(256) void k_norm(const float* __restrict__ P,
                                              f16* __restrict__ A16, f16* __restrict__ At16) {
  __shared__ float red[256];
  __shared__ float rsum, csum;
  int v = blockIdx.x, tid = threadIdx.x;
  if (v >= NN) {
    for (int j = tid; j < NPAD; j += 256) { A16[(size_t)v * NPAD + j] = (f16)0.f; At16[(size_t)v * NPAD + j] = (f16)0.f; }
    return;
  }
  float rs = 0.f, cs = 0.f;
  for (int j = tid; j < NN; j += 256) { rs += P[(size_t)v * NPAD + j]; cs += P[(size_t)j * NPAD + v]; }
  red[tid] = rs; __syncthreads();
  for (int off = 128; off > 0; off >>= 1) { if (tid < off) red[tid] += red[tid + off]; __syncthreads(); }
  if (tid == 0) rsum = red[0];
  __syncthreads();
  red[tid] = cs; __syncthreads();
  for (int off = 128; off > 0; off >>= 1) { if (tid < off) red[tid] += red[tid + off]; __syncthreads(); }
  if (tid == 0) csum = red[0];
  __syncthreads();
  float rinv = 1.f / (1.f + rsum), cinv = 1.f / (1.f + csum);
  for (int j = tid; j < NPAD; j += 256) {
    float pv = (j < NN) ? P[(size_t)v * NPAD + j] : 0.f;
    float pt = (j < NN) ? P[(size_t)j * NPAD + v] : 0.f;
    float d  = (j == v) ? 1.f : 0.f;
    A16[(size_t)v * NPAD + j]  = (f16)((j < NN) ? ((pv + d) * rinv) : 0.f);
    At16[(size_t)v * NPAD + j] = (f16)((j < NN) ? ((pt + d) * cinv) : 0.f);
  }
}

// ---------------- zero the node-padding columns (n in [NN,NPAD)) of both f16 feature buffers ----------------
__global__ void k_zpad(f16* __restrict__ xg16, f16* __restrict__ h16w) {
  int gid = blockIdx.x * 256 + threadIdx.x;
  const int per = 256 * TMAXX * (NPAD - NN);
  if (gid >= 2 * per) return;
  f16* buf = (gid < per) ? xg16 : h16w;
  int g = gid % per;
  int n = g % (NPAD - NN); int t = (g / (NPAD - NN)) % TMAXX; int slab = g / ((NPAD - NN) * TMAXX);
  buf[((size_t)slab * TMAXX + t) * NPAD + NN + n] = (f16)0.f;
}

// ---------------- start_conv (1x1, C=2 -> RC=8) ----------------
__global__ void k_start(const float* __restrict__ in, const float* __restrict__ sw,
                        const float* __restrict__ sb, float* __restrict__ x) {
  int gid = blockIdx.x * 256 + threadIdx.x;
  const int T = 168;
  if (gid >= BB * RCH * NN * T) return;
  int t = gid % T; int n = (gid / T) % NN; int o = (gid / (T * NN)) % RCH; int b = gid / (T * NN * RCH);
  float v = sb[o];
  v += in[((size_t)(b * 2 + 0) * NN + n) * T + t] * sw[o * 2 + 0];
  v += in[((size_t)(b * 2 + 1) * NN + n) * T + t] * sw[o * 2 + 1];
  x[gid] = v;
}

// ---------------- inception (4 branches) + tanh*sigmoid gate -> xg16 (f16, [slab][t][n]) ----------------
__global__ __launch_bounds__(256) void k_incep(const float* __restrict__ x, f16* __restrict__ xg16,
    const float* fw2, const float* fb2, const float* fw4, const float* fb4,
    const float* fw6, const float* fb6, const float* fw8, const float* fb8,
    const float* gw2, const float* gb2, const float* gw4, const float* gb4,
    const float* gw6, const float* gb6, const float* gw8, const float* gb8,
    int l, int Tin, int Tout) {
  __shared__ float wf[1280], wg[1280], bfs[32], bgs[32];
  const int tid = threadIdx.x;
  const float* fws[4] = {fw2, fw4, fw6, fw8};
  const float* fbs[4] = {fb2, fb4, fb6, fb8};
  const float* gws[4] = {gw2, gw4, gw6, gw8};
  const float* gbs[4] = {gb2, gb4, gb6, gb8};
  const int KS[4]  = {2, 4, 6, 8};
  const int OBR[4] = {0, 128, 384, 768};
  for (int br = 0; br < 4; ++br) {
    int sz = 64 * KS[br];
    const float* fwp = fws[br] + l * sz;
    const float* gwp = gws[br] + l * sz;
    for (int idx = tid; idx < sz; idx += 256) { wf[OBR[br] + idx] = fwp[idx]; wg[OBR[br] + idx] = gwp[idx]; }
  }
  for (int idx = tid; idx < 32; idx += 256) {
    bfs[idx] = fbs[idx >> 3][l * 8 + (idx & 7)];
    bgs[idx] = gbs[idx >> 3][l * 8 + (idx & 7)];
  }
  __syncthreads();
  int gid = blockIdx.x * 256 + tid;
  int total = BB * NN * Tout;
  if (gid >= total) return;
  int n = gid % NN; int t = (gid / NN) % Tout; int b = gid / (NN * Tout);
  float xv[8][8];
#pragma unroll
  for (int ic = 0; ic < 8; ++ic) {
    const float* xp = x + ((size_t)(b * RCH + ic) * NN + n) * Tin + t;
#pragma unroll
    for (int j = 0; j < 8; ++j) xv[ic][j] = xp[j];
  }
#pragma unroll
  for (int br = 0; br < 4; ++br) {
    const int K = KS[br];
    const int off = 8 - K;   // branch-k output truncated to last Tin-7 samples
#pragma unroll
    for (int oc = 0; oc < 8; ++oc) {
      float f = bfs[br * 8 + oc], g = bgs[br * 8 + oc];
#pragma unroll
      for (int ic = 0; ic < 8; ++ic) {
#pragma unroll
        for (int j = 0; j < 8; ++j) {
          if (j < K) {
            float xvv = xv[ic][off + j];
            f += xvv * wf[OBR[br] + (oc * 8 + ic) * K + j];
            g += xvv * wg[OBR[br] + (oc * 8 + ic) * K + j];
          }
        }
      }
      float val = tanhf(f) * (1.f / (1.f + expf(-g)));
      int c = br * 8 + oc;
      xg16[((size_t)(b * CCH + c) * TMAXX + t) * NPAD + n] = (f16)val;
    }
  }
}

// ---------------- xnew = residual + biases + h0 projection through (g1_w+g2_w); xnew is [b][o][t][n] ----------------
__global__ __launch_bounds__(256) void k_init(const float* __restrict__ x, const f16* __restrict__ xg16,
    float* __restrict__ xnew, const float* __restrict__ g1w, const float* __restrict__ g1b,
    const float* __restrict__ g2w, const float* __restrict__ g2b, int l, int Tin, int Tout) {
  __shared__ float W[256];
  __shared__ float bsh[8];
  int tid = threadIdx.x;
  for (int idx = tid; idx < 256; idx += 256) {
    int o = idx >> 5, c = idx & 31;
    W[idx] = g1w[(l * 8 + o) * 96 + c] + g2w[(l * 8 + o) * 96 + c];
  }
  if (tid < 8) bsh[tid] = g1b[l * 8 + tid] + g2b[l * 8 + tid];
  __syncthreads();
  int gid = blockIdx.x * 256 + tid;
  int total = BB * NN * Tout;
  if (gid >= total) return;
  int n = gid % NN; int t = (gid / NN) % Tout; int b = gid / (NN * Tout);
  float hv[32];
#pragma unroll
  for (int c = 0; c < 32; ++c)
    hv[c] = (float)xg16[((size_t)(b * CCH + c) * TMAXX + t) * NPAD + n];
#pragma unroll
  for (int o = 0; o < 8; ++o) {
    float acc = bsh[o] + x[((size_t)(b * RCH + o) * NN + n) * Tin + (Tin - Tout) + t];
#pragma unroll
    for (int c = 0; c < 32; ++c) acc += hv[c] * W[o * 32 + c];
    xnew[((size_t)(b * RCH + o) * TMAXX + t) * NN + n] = acc;
  }
}

// ---------------- WMMA GEMM: D[t,v] = sum_w H[t,w] * M[w,v] with M[w,v] = Amat[v*NPAD+w] ----------------
// H is f16 [slab][TMAXX][NPAD] (w contiguous). One wave per 32(t) x 64(v) tile;
// A-operand = H rows (t), B-operand = Amat rows (v); all loads unconditional b128s.
__global__ __launch_bounds__(32) void k_gemm(const f16* __restrict__ Amat, const f16* __restrict__ H,
                                             float* __restrict__ D, int Tcur) {
  const int lane = threadIdx.x;
  const int v0 = blockIdx.x * 64;
  const int t0 = blockIdx.y * 32;
  const int slab = blockIdx.z;
  const int col = lane & 15;
  const int hi  = lane >> 4;
  const f16* Hs = H + (size_t)slab * TMAXX * NPAD;
  // A-operand rows (t). Out-of-range rows clamped to 0 (in-bounds garbage; stores guarded).
  int tA0 = t0 + col;       if (tA0 >= Tcur) tA0 = 0;
  int tA1 = t0 + 16 + col;  if (tA1 >= Tcur) tA1 = 0;
  const f16* hp0 = Hs + (size_t)tA0 * NPAD + hi * 16;
  const f16* hp1 = Hs + (size_t)tA1 * NPAD + hi * 16;
  // B-operand row pointers: lane holds column v; K halves contiguous at Amat[v*NPAD + kc + hi*16]
  const f16* bp0 = Amat + (size_t)(v0 + 0  + col) * NPAD + hi * 16;
  const f16* bp1 = Amat + (size_t)(v0 + 16 + col) * NPAD + hi * 16;
  const f16* bp2 = Amat + (size_t)(v0 + 32 + col) * NPAD + hi * 16;
  const f16* bp3 = Amat + (size_t)(v0 + 48 + col) * NPAD + hi * 16;
  v8f acc[8];
#pragma unroll
  for (int q = 0; q < 8; ++q)
#pragma unroll
    for (int r = 0; r < 8; ++r) acc[q][r] = 0.f;

#define LOAD_A(dst, base, kk)                                            \
  {                                                                      \
    v8h lo_ = *(const v8h*)((base) + (kk));                              \
    v8h hi_ = *(const v8h*)((base) + (kk) + 8);                          \
    _Pragma("unroll") for (int q_ = 0; q_ < 8; ++q_) { (dst)[q_] = lo_[q_]; (dst)[8 + q_] = hi_[q_]; } \
  }
#define LOAD_B(dst, base, kk)                                            \
  {                                                                      \
    v8h lo_ = *(const v8h*)((base) + (kk));                              \
    v8h hi_ = *(const v8h*)((base) + (kk) + 8);                          \
    _Pragma("unroll") for (int q_ = 0; q_ < 8; ++q_) { (dst)[q_] = lo_[q_]; (dst)[8 + q_] = hi_[q_]; } \
  }

  for (int kc = 0; kc < NPAD; kc += 64) {
    // stage 0 fragments
    v16h a0, a1, b0, b1, b2, b3;
    LOAD_A(a0, hp0, kc); LOAD_A(a1, hp1, kc);
    LOAD_B(b0, bp0, kc); LOAD_B(b1, bp1, kc);
    LOAD_B(b2, bp2, kc); LOAD_B(b3, bp3, kc);
    // stage 1 fragments (loads issue before stage-0 WMMAs consume)
    v16h c0, c1, d0, d1, d2, d3;
    LOAD_A(c0, hp0, kc + 32); LOAD_A(c1, hp1, kc + 32);
    LOAD_B(d0, bp0, kc + 32); LOAD_B(d1, bp1, kc + 32);
    LOAD_B(d2, bp2, kc + 32); LOAD_B(d3, bp3, kc + 32);
    acc[0] = __builtin_amdgcn_wmma_f32_16x16x32_f16(false, a0, false, b0, (short)0, acc[0], false, false);
    acc[1] = __builtin_amdgcn_wmma_f32_16x16x32_f16(false, a0, false, b1, (short)0, acc[1], false, false);
    acc[2] = __builtin_amdgcn_wmma_f32_16x16x32_f16(false, a0, false, b2, (short)0, acc[2], false, false);
    acc[3] = __builtin_amdgcn_wmma_f32_16x16x32_f16(false, a0, false, b3, (short)0, acc[3], false, false);
    acc[4] = __builtin_amdgcn_wmma_f32_16x16x32_f16(false, a1, false, b0, (short)0, acc[4], false, false);
    acc[5] = __builtin_amdgcn_wmma_f32_16x16x32_f16(false, a1, false, b1, (short)0, acc[5], false, false);
    acc[6] = __builtin_amdgcn_wmma_f32_16x16x32_f16(false, a1, false, b2, (short)0, acc[6], false, false);
    acc[7] = __builtin_amdgcn_wmma_f32_16x16x32_f16(false, a1, false, b3, (short)0, acc[7], false, false);
    acc[0] = __builtin_amdgcn_wmma_f32_16x16x32_f16(false, c0, false, d0, (short)0, acc[0], false, false);
    acc[1] = __builtin_amdgcn_wmma_f32_16x16x32_f16(false, c0, false, d1, (short)0, acc[1], false, false);
    acc[2] = __builtin_amdgcn_wmma_f32_16x16x32_f16(false, c0, false, d2, (short)0, acc[2], false, false);
    acc[3] = __builtin_amdgcn_wmma_f32_16x16x32_f16(false, c0, false, d3, (short)0, acc[3], false, false);
    acc[4] = __builtin_amdgcn_wmma_f32_16x16x32_f16(false, c1, false, d0, (short)0, acc[4], false, false);
    acc[5] = __builtin_amdgcn_wmma_f32_16x16x32_f16(false, c1, false, d1, (short)0, acc[5], false, false);
    acc[6] = __builtin_amdgcn_wmma_f32_16x16x32_f16(false, c1, false, d2, (short)0, acc[6], false, false);
    acc[7] = __builtin_amdgcn_wmma_f32_16x16x32_f16(false, c1, false, d3, (short)0, acc[7], false, false);
  }
#undef LOAD_A
#undef LOAD_B
  // D fragment: lane col = v, VGPR r -> row m = r + 8*hi (m is the t dimension)
  float* Ds = D + (size_t)slab * TMAXX * NPAD;
#pragma unroll
  for (int ai = 0; ai < 2; ++ai)
#pragma unroll
    for (int vi = 0; vi < 4; ++vi)
#pragma unroll
      for (int r = 0; r < 8; ++r) {
        int t = t0 + ai * 16 + r + 8 * hi;
        if (t < Tcur)
          Ds[(size_t)t * NPAD + v0 + vi * 16 + col] = acc[ai * 4 + vi][r];
      }
}

// ---------------- post-hop: h = PA*x0 + (1-PA)*Ah; re-encode f16; accumulate 96->8 projection ----------------
__global__ __launch_bounds__(256) void k_post(const float* __restrict__ Ah, const f16* __restrict__ xg16,
    f16* __restrict__ hout, float* __restrict__ xnew, const float* __restrict__ Wsrc,
    int l, int colbase, int Tout, int writeH) {
  __shared__ float W[256];
  int tid = threadIdx.x;
  for (int idx = tid; idx < 256; idx += 256) {
    int o = idx >> 5, c = idx & 31;
    W[idx] = Wsrc[(l * 8 + o) * 96 + colbase + c];
  }
  __syncthreads();
  int gid = blockIdx.x * 256 + tid;
  int total = BB * NN * Tout;
  if (gid >= total) return;
  int n = gid % NN; int t = (gid / NN) % Tout; int b = gid / (NN * Tout);
  float hv[32];
#pragma unroll
  for (int c = 0; c < 32; ++c) {
    size_t pidx = ((size_t)(b * CCH + c) * TMAXX + t) * NPAD + n;
    float x0 = (float)xg16[pidx];
    float av = Ah[pidx];
    float h = GPA * x0 + (1.f - GPA) * av;
    hv[c] = h;
    if (writeH) hout[pidx] = (f16)h;
  }
#pragma unroll
  for (int o = 0; o < 8; ++o) {
    size_t oi = ((size_t)(b * RCH + o) * TMAXX + t) * NN + n;
    float acc = xnew[oi];
#pragma unroll
    for (int c = 0; c < 32; ++c) acc += hv[c] * W[o * 32 + c];
    xnew[oi] = acc;
  }
}

// ---------------- layernorm over (C,N,T) per batch ----------------
__global__ void k_lnzero(float* red) { if (threadIdx.x < 16) red[threadIdx.x] = 0.f; }

__global__ __launch_bounds__(256) void k_lnred(const float* __restrict__ xnew, float* red, int Tout) {
  __shared__ float s1[256], s2[256];
  int b = blockIdx.y, tid = threadIdx.x;
  int M = RCH * NN * Tout;
  float a1 = 0.f, a2 = 0.f;
  for (int m = blockIdx.x * 256 + tid; m < M; m += gridDim.x * 256) {
    int n = m % NN; int t = (m / NN) % Tout; int c = m / (NN * Tout);
    float v = xnew[((size_t)(b * RCH + c) * TMAXX + t) * NN + n];
    a1 += v; a2 += v * v;
  }
  s1[tid] = a1; s2[tid] = a2; __syncthreads();
  for (int off = 128; off > 0; off >>= 1) {
    if (tid < off) { s1[tid] += s1[tid + off]; s2[tid] += s2[tid + off]; }
    __syncthreads();
  }
  if (tid == 0) { atomicAdd(&red[2 * b], s1[0]); atomicAdd(&red[2 * b + 1], s2[0]); }
}

__global__ __launch_bounds__(256) void k_lnapp(const float* __restrict__ xnew, float* __restrict__ xout,
    const float* __restrict__ nw, const float* __restrict__ nb, const float* __restrict__ red, int Tout) {
  int gid = blockIdx.x * 256 + threadIdx.x;
  int total = BB * RCH * NN * Tout;
  if (gid >= total) return;
  int t = gid % Tout; int n = (gid / Tout) % NN; int c = (gid / (Tout * NN)) % RCH; int b = gid / (Tout * NN * RCH);
  float M = (float)(RCH * NN * Tout);
  float mu = red[2 * b] / M;
  float var = red[2 * b + 1] / M - mu * mu;
  float inv = rsqrtf(var + 1e-5f);
  float v = xnew[((size_t)(b * RCH + c) * TMAXX + t) * NN + n];
  int wi = (c * NN + n) * Tout + t;
  xout[gid] = (v - mu) * inv * nw[wi] + nb[wi];
}

// ---------------- adaptive avg pool T->12, output [B,C,12,N] ----------------
__global__ void k_pool(const float* __restrict__ x, float* __restrict__ out, int T) {
  int gid = blockIdx.x * 256 + threadIdx.x;
  if (gid >= BB * RCH * 12 * NN) return;
  int n = gid % NN; int s = (gid / NN) % 12; int c = (gid / (NN * 12)) % RCH; int b = gid / (NN * 12 * RCH);
  int st = (s * T) / 12;
  int en = ((s + 1) * T + 11) / 12;
  float sum = 0.f;
  for (int t = st; t < en; ++t) sum += x[((size_t)(b * RCH + c) * NN + n) * T + t];
  out[gid] = sum / (float)(en - st);
}

extern "C" void kernel_launch(void* const* d_in, const int* in_sizes, int n_in,
                              void* d_out, int out_size, void* d_ws, size_t ws_size,
                              hipStream_t stream) {
  (void)in_sizes; (void)n_in; (void)out_size; (void)ws_size;
  const float* inp  = (const float*)d_in[0];
  const float* cool = (const float*)d_in[1];
  const float* noise= (const float*)d_in[2];
  const float* emb1 = (const float*)d_in[3];
  const float* emb2 = (const float*)d_in[4];
  const float* l1w  = (const float*)d_in[5];
  const float* l1b  = (const float*)d_in[6];
  const float* l2w  = (const float*)d_in[7];
  const float* l2b  = (const float*)d_in[8];
  const float* sw   = (const float*)d_in[9];
  const float* sb   = (const float*)d_in[10];
  // setup_inputs dict order: fw2,fb2,fw4,fb4,fw6,fb6,fw8,fb8, gw2,gb2,...
  const float* fw[4] = {(const float*)d_in[11], (const float*)d_in[13], (const float*)d_in[15], (const float*)d_in[17]};
  const float* fb[4] = {(const float*)d_in[12], (const float*)d_in[14], (const float*)d_in[16], (const float*)d_in[18]};
  const float* gw[4] = {(const float*)d_in[19], (const float*)d_in[21], (const float*)d_in[23], (const float*)d_in[25]};
  const float* gb[4] = {(const float*)d_in[20], (const float*)d_in[22], (const float*)d_in[24], (const float*)d_in[26]};
  const float* g1w = (const float*)d_in[27];
  const float* g1b = (const float*)d_in[28];
  const float* g2w = (const float*)d_in[29];
  const float* g2b = (const float*)d_in[30];
  const float* nw[3] = {(const float*)d_in[31], (const float*)d_in[33], (const float*)d_in[35]};
  const float* nb[3] = {(const float*)d_in[32], (const float*)d_in[34], (const float*)d_in[36]};

  char* ws = (char*)d_ws;
  float* v1   = (float*)(ws + 0);           //    160000 B
  float* v2   = (float*)(ws + 160000);      //    160000 B
  float* P    = (float*)(ws + 320000);      //   4194304 B  [1024x1024] f32
  f16*  A16   = (f16*)  (ws + 4514304);     //   2097152 B
  f16*  At16  = (f16*)  (ws + 6611456);     //   2097152 B
  float* x    = (float*)(ws + 8708608);     //  43008000 B  [8,8,1000,168] f32
  float* xnew = (float*)(ws + 51716608);    //  41216000 B  [8,8,161,1000] f32
  f16*  xg16  = (f16*)  (ws + 92932608);    //  84410368 B  [256,161,1024] f16
  f16*  h16w  = (f16*)  (ws + 177342976);   //  84410368 B
  float* Ah   = (float*)(ws + 261753344);   // 168820736 B [256,161,1024] f32
  float* red  = (float*)(ws + 430574080);   //        64 B

  k_v12<<<dim3((NN * NDIM + 255) / 256), 256, 0, stream>>>(emb1, emb2, l1w, l1b, l2w, l2b, v1, v2);
  k_adj<<<dim3(NPAD), 256, 0, stream>>>(v1, v2, noise, cool, P);
  k_norm<<<dim3(NPAD), 256, 0, stream>>>(P, A16, At16);
  {
    int tot = 2 * 256 * TMAXX * (NPAD - NN);
    k_zpad<<<dim3((tot + 255) / 256), 256, 0, stream>>>(xg16, h16w);
  }
  {
    int tot = BB * RCH * NN * 168;
    k_start<<<dim3((tot + 255) / 256), 256, 0, stream>>>(inp, sw, sb, x);
  }

  int Tin = 168;
  for (int l = 0; l < 3; ++l) {
    int Tout = Tin - 7;
    int totBNT = BB * NN * Tout;
    dim3 gBNT((totBNT + 255) / 256);
    k_incep<<<gBNT, 256, 0, stream>>>(x, xg16,
        fw[0], fb[0], fw[1], fb[1], fw[2], fb[2], fw[3], fb[3],
        gw[0], gb[0], gw[1], gb[1], gw[2], gb[2], gw[3], gb[3], l, Tin, Tout);
    k_init<<<gBNT, 256, 0, stream>>>(x, xg16, xnew, g1w, g1b, g2w, g2b, l, Tin, Tout);

    dim3 gG(NPAD / 64, (Tout + 31) / 32, BB * CCH);
    // mixprop 1 (A)
    k_gemm<<<gG, 32, 0, stream>>>(A16, xg16, Ah, Tout);
    k_post<<<gBNT, 256, 0, stream>>>(Ah, xg16, h16w, xnew, g1w, l, 32, Tout, 1);
    k_gemm<<<gG, 32, 0, stream>>>(A16, h16w, Ah, Tout);
    k_post<<<gBNT, 256, 0, stream>>>(Ah, xg16, h16w, xnew, g1w, l, 64, Tout, 0);
    // mixprop 2 (A^T)
    k_gemm<<<gG, 32, 0, stream>>>(At16, xg16, Ah, Tout);
    k_post<<<gBNT, 256, 0, stream>>>(Ah, xg16, h16w, xnew, g2w, l, 32, Tout, 1);
    k_gemm<<<gG, 32, 0, stream>>>(At16, h16w, Ah, Tout);
    k_post<<<gBNT, 256, 0, stream>>>(Ah, xg16, h16w, xnew, g2w, l, 64, Tout, 0);
    // layernorm
    k_lnzero<<<1, 32, 0, stream>>>(red);
    k_lnred<<<dim3(128, BB), 256, 0, stream>>>(xnew, red, Tout);
    int totA = BB * RCH * NN * Tout;
    k_lnapp<<<dim3((totA + 255) / 256), 256, 0, stream>>>(xnew, x, nw[l], nb[l], red, Tout);
    Tin = Tout;
  }
  k_pool<<<dim3((BB * RCH * 12 * NN + 255) / 256), 256, 0, stream>>>(x, (float*)d_out, Tin);
}
